// EncDec_16286515986717
// MI455X (gfx1250) — matrix operand order repaired and checked by reading, not compiled, not run on hardware
//
#include <hip/hip_runtime.h>

// ---------------- Problem constants (from reference) ----------------
#define B_ 64
#define T_ 128
#define F_ 2048
#define H_ 1024
#define C_ 22
#define G4H_ 4096        // 4*H
#define BT_ 8192         // B*T

typedef unsigned short ushort_t;
typedef __attribute__((ext_vector_type(16))) __bf16 v16bf;
typedef __attribute__((ext_vector_type(8)))  float  v8f;

union Frag16 { v16bf v; unsigned int u[8]; };

#define DEV_ static __device__ __forceinline__

DEV_ ushort_t f2bf(float f) {
  unsigned int u = __float_as_uint(f);
  u += 0x7FFFu + ((u >> 16) & 1u);       // round-to-nearest-even
  return (ushort_t)(u >> 16);
}
DEV_ float bf2f(ushort_t h) { return __uint_as_float(((unsigned int)h) << 16); }
DEV_ float sigm(float x) { return 1.0f / (1.0f + __expf(-x)); }

// =====================================================================
// Generic bf16 WMMA GEMM:  C[M,N] = A[M,K] * W[N,K]^T  (+addend)(+bias)
//   A, W bf16 row-major (lda/ldw in elements); output fp32 C and/or bf16 Cb.
//   Block tile 64(M) x 128(N), BK=32; grid = (N/128, M/64); 256 threads.
//   8 waves arranged 4(M) x 2(N): each wave computes 16x64 = 4 WMMA tiles,
//   reusing ONE A fragment across 4 B fragments (2x the math per LDS byte
//   vs a 16x32 wave tile). All dims are tile multiples -> no bounds checks,
//   EXEC stays all-ones around WMMA as the ISA requires.
// =====================================================================
__global__ __launch_bounds__(256)
void gemm_wmma_bf16(const ushort_t* __restrict__ A, int lda,
                    const ushort_t* __restrict__ W, int ldw,
                    const float* __restrict__ addend, int ld_add,
                    const float* __restrict__ bias,
                    float* __restrict__ C, ushort_t* __restrict__ Cb,
                    int ldc, int K)
{
  const int tid  = threadIdx.x;
  const int lane = tid & 31;
  const int wid  = tid >> 5;
  const int wm   = wid & 3;          // 0..3  (16-row slab)
  const int wn   = wid >> 2;         // 0..1  (64-col slab)
  const int half = lane >> 4;        // 0/1
  const int l16  = lane & 15;
  const int m0   = blockIdx.y * 64;
  const int n0   = blockIdx.x * 128;

  __shared__ ushort_t As[64 * 34];    // 64 rows  x 32 k (stride 34 pad)
  __shared__ ushort_t Ws[128 * 34];   // 128 rows x 32 k

  v8f acc[4] = {v8f{}, v8f{}, v8f{}, v8f{}};

  const int row = tid >> 2;          // 0..63  cooperative-load row
  const int cg  = (tid & 3) * 8;     // 0/8/16/24 element group

  const ushort_t* ap  = A + (size_t)(m0 + row) * lda + cg;
  const ushort_t* wp0 = W + (size_t)(n0 + row) * ldw + cg;
  const ushort_t* wp1 = W + (size_t)(n0 + 64 + row) * ldw + cg;

  for (int k0 = 0; k0 < K; k0 += 32) {
    // ---- cooperative stage: 64x32 A-tile, 128x32 W-tile into LDS ----
    uint4 av = *(const uint4*)(ap  + k0);
    uint4 w0 = *(const uint4*)(wp0 + k0);
    uint4 w1 = *(const uint4*)(wp1 + k0);
    unsigned int* as_ = (unsigned int*)&As[row * 34 + cg];
    unsigned int* wl_ = (unsigned int*)&Ws[row * 34 + cg];
    unsigned int* wh_ = (unsigned int*)&Ws[(64 + row) * 34 + cg];
    as_[0] = av.x; as_[1] = av.y; as_[2] = av.z; as_[3] = av.w;
    wl_[0] = w0.x; wl_[1] = w0.y; wl_[2] = w0.z; wl_[3] = w0.w;
    wh_[0] = w1.x; wh_[1] = w1.y; wh_[2] = w1.z; wh_[3] = w1.w;
    if (k0 + 32 < K) {               // hint next tiles -> global_prefetch_b8
      __builtin_prefetch(ap  + k0 + 32, 0, 1);
      __builtin_prefetch(wp0 + k0 + 32, 0, 1);
      __builtin_prefetch(wp1 + k0 + 32, 0, 1);
    }
    __syncthreads();

    // ---- A fragment (16x32, MxK), loaded once, reused 4x ----
    Frag16 fa;
    {
      const ushort_t* b = &As[(wm * 16 + l16) * 34];
      const int kk = half * 8;
      fa.u[0] = *(const unsigned int*)&b[kk + 0];
      fa.u[1] = *(const unsigned int*)&b[kk + 2];
      fa.u[2] = *(const unsigned int*)&b[kk + 4];
      fa.u[3] = *(const unsigned int*)&b[kk + 6];
      fa.u[4] = *(const unsigned int*)&b[16 + kk + 0];
      fa.u[5] = *(const unsigned int*)&b[16 + kk + 2];
      fa.u[6] = *(const unsigned int*)&b[16 + kk + 4];
      fa.u[7] = *(const unsigned int*)&b[16 + kk + 6];
    }
    // ---- 4 B fragments (32x16, KxN) for the wave's 64-col slab ----
    Frag16 fb[4];
#pragma unroll
    for (int sub = 0; sub < 4; ++sub) {
      const ushort_t* b = &Ws[(wn * 64 + sub * 16 + l16) * 34 + half * 8];
#pragma unroll
      for (int j = 0; j < 8; ++j) fb[sub].u[j] = *(const unsigned int*)&b[2 * j + (j >= 4 ? 8 : 0)];
    }
#pragma unroll
    for (int sub = 0; sub < 4; ++sub)
      acc[sub] = __builtin_amdgcn_wmma_f32_16x16x32_bf16(
          false, fa.v, false, fb[sub].v, (short)0, acc[sub], false, false);
    __syncthreads();
  }

  // ---- epilogue: C/D layout -> lane l16 = col, vgpr r + 8*half = row ----
#pragma unroll
  for (int sub = 0; sub < 4; ++sub) {
#pragma unroll
    for (int r = 0; r < 8; ++r) {
      const int m = m0 + wm * 16 + half * 8 + r;
      const int n = n0 + wn * 64 + sub * 16 + l16;
      float v = acc[sub][r];
      if (bias)   v += bias[n];
      if (addend) v += addend[(size_t)m * ld_add + n];
      if (C)  C[(size_t)m * ldc + n]  = v;
      if (Cb) Cb[(size_t)m * ldc + n] = f2bf(v);
    }
  }
}

// =====================================================================
// Elementwise / small kernels
// =====================================================================
__global__ void cvt_f32_bf16(const float* __restrict__ src, int sld,
                             ushort_t* __restrict__ dst, int dld,
                             int rows, int cols)
{
  long n = (long)rows * cols;
  for (long i = (long)blockIdx.x * blockDim.x + threadIdx.x; i < n;
       i += (long)gridDim.x * blockDim.x) {
    int r = (int)(i / cols), c = (int)(i % cols);
    dst[(size_t)r * dld + c] = f2bf(src[(size_t)r * sld + c]);
  }
}

__global__ void vec_add(const float* a, const float* b, float* o, int n)
{
  int i = blockIdx.x * blockDim.x + threadIdx.x;
  if (i < n) o[i] = a[i] + b[i];
}

__global__ void zero_u32(unsigned int* p, long n)
{
  for (long i = (long)blockIdx.x * blockDim.x + threadIdx.x; i < n;
       i += (long)gridDim.x * blockDim.x) p[i] = 0u;
}

// gates G[b, 0:4H] in i,f,g,o order -> LSTM cell update (c fp32 in-place,
// h written bf16; optionally scattered into h_ts at time t)
__global__ void lstm_cell_k(const float* __restrict__ G,
                            float* __restrict__ c,
                            ushort_t* __restrict__ h_out,
                            ushort_t* __restrict__ hts_out, int t)
{
  int idx = blockIdx.x * blockDim.x + threadIdx.x;   // B_*H_ threads
  int b = idx >> 10, hh = idx & (H_ - 1);
  const float* g = G + (size_t)b * G4H_;
  float gi = g[hh], gf = g[H_ + hh], gg = g[2 * H_ + hh], go = g[3 * H_ + hh];
  float cn = sigm(gf) * c[idx] + sigm(gi) * tanhf(gg);
  float hn = sigm(go) * tanhf(cn);
  c[idx] = cn;
  ushort_t hb = f2bf(hn);
  h_out[idx] = hb;
  if (hts_out) hts_out[((size_t)b * T_ + t) * H_ + hh] = hb;
}

// logits[b,t'] = scale * <h[b,:], proj[b,t',:]>  (one wave per (b,t'))
__global__ void attn_logits_k(const ushort_t* __restrict__ h,
                              const ushort_t* __restrict__ proj,
                              float* __restrict__ logits, float scale)
{
  int gw = (blockIdx.x * blockDim.x + threadIdx.x) >> 5;  // wave id, B_*T_
  int lane = threadIdx.x & 31;
  int b = gw / T_, tt = gw % T_;
  const ushort_t* hv = h + (size_t)b * H_;
  const ushort_t* pv = proj + ((size_t)b * T_ + tt) * H_;
  float s = 0.f;
  for (int k = lane; k < H_; k += 32) s += bf2f(hv[k]) * bf2f(pv[k]);
#pragma unroll
  for (int m = 16; m > 0; m >>= 1) s += __shfl_xor(s, m, 32);
  if (lane == 0) logits[(size_t)b * T_ + tt] = s * scale;
}

__global__ void softmax_rows_k(const float* __restrict__ logits,
                               float* __restrict__ aw,
                               float* __restrict__ aw_out)
{
  int b = blockIdx.x, t = threadIdx.x;        // T_ threads
  __shared__ float red[T_];
  float v = logits[(size_t)b * T_ + t];
  red[t] = v; __syncthreads();
  for (int s = T_ / 2; s > 0; s >>= 1) {
    if (t < s) red[t] = fmaxf(red[t], red[t + s]);
    __syncthreads();
  }
  float mx = red[0]; __syncthreads();
  float e = __expf(v - mx);
  red[t] = e; __syncthreads();
  for (int s = T_ / 2; s > 0; s >>= 1) {
    if (t < s) red[t] += red[t + s];
    __syncthreads();
  }
  float r = e / red[0];
  aw[(size_t)b * T_ + t] = r;
  if (aw_out) aw_out[(size_t)b * T_ + t] = r;
}

// attn[b,h] = sum_t aw[b,t] * proj[b,t,h]  (bf16 out)
__global__ void attn_sum_k(const float* __restrict__ aw,
                           const ushort_t* __restrict__ proj,
                           ushort_t* __restrict__ attn)
{
  int idx = blockIdx.x * blockDim.x + threadIdx.x;   // B_*H_
  int b = idx >> 10, hh = idx & (H_ - 1);
  float s = 0.f;
  for (int tt = 0; tt < T_; ++tt)
    s += aw[(size_t)b * T_ + tt] * bf2f(proj[((size_t)b * T_ + tt) * H_ + hh]);
  attn[idx] = f2bf(s);
}

// scores[b,t,cc] = <h[b,:], Wc[cc,:]> + bc[cc]
__global__ void classifier_k(const ushort_t* __restrict__ h,
                             const float* __restrict__ Wc,
                             const float* __restrict__ bc,
                             float* __restrict__ out, int t)
{
  int idx = blockIdx.x * blockDim.x + threadIdx.x;   // B_*C_
  if (idx >= B_ * C_) return;
  int b = idx / C_, cc = idx % C_;
  float s = bc[cc];
  const ushort_t* hv = h + (size_t)b * H_;
  const float* w = Wc + (size_t)cc * H_;
  for (int k = 0; k < H_; ++k) s += bf2f(hv[k]) * w[k];
  out[((size_t)b * T_ + t) * C_ + cc] = s;
}

// =====================================================================
// Host orchestration
// =====================================================================
extern "C" void kernel_launch(void* const* d_in, const int* in_sizes, int n_in,
                              void* d_out, int out_size, void* d_ws, size_t ws_size,
                              hipStream_t stream)
{
  (void)in_sizes; (void)n_in; (void)out_size; (void)ws_size;
  const float* x        = (const float*)d_in[0];
  const float* enc_Wih  = (const float*)d_in[1];
  const float* enc_Whh  = (const float*)d_in[2];
  const float* enc_bih  = (const float*)d_in[3];
  const float* enc_bhh  = (const float*)d_in[4];
  const float* We2d     = (const float*)d_in[5];
  const float* be2d     = (const float*)d_in[6];
  const float* Wd2e     = (const float*)d_in[7];
  const float* bd2e     = (const float*)d_in[8];
  const float* dec_Wih  = (const float*)d_in[9];
  const float* dec_Whh  = (const float*)d_in[10];
  const float* dec_bih  = (const float*)d_in[11];
  const float* dec_bhh  = (const float*)d_in[12];
  const float* Wc       = (const float*)d_in[13];
  const float* bc       = (const float*)d_in[14];
  float* out_scores = (float*)d_out;                        // [B,T,C]
  float* out_aw     = (float*)d_out + (size_t)B_ * T_ * C_; // [B,T]

  // ---- carve workspace ----
  char* base = (char*)d_ws; size_t off = 0;
  auto alloc = [&](size_t bytes) -> void* {
    void* p = base + off;
    off = (off + bytes + 255) & ~(size_t)255;
    return p;
  };
  ushort_t* xb      = (ushort_t*)alloc((size_t)BT_ * F_ * 2);
  ushort_t* Wih_b   = (ushort_t*)alloc((size_t)G4H_ * F_ * 2);
  ushort_t* Whh_b   = (ushort_t*)alloc((size_t)G4H_ * H_ * 2);
  ushort_t* We2d_b  = (ushort_t*)alloc((size_t)H_ * H_ * 2);
  ushort_t* Wd2e_b  = (ushort_t*)alloc((size_t)H_ * H_ * 2);
  ushort_t* dWx_b   = (ushort_t*)alloc((size_t)G4H_ * F_ * 2);
  ushort_t* dWa_b   = (ushort_t*)alloc((size_t)G4H_ * H_ * 2);
  ushort_t* dWhh_b  = (ushort_t*)alloc((size_t)G4H_ * H_ * 2);
  float*    encb    = (float*)alloc((size_t)G4H_ * 4);
  float*    decb    = (float*)alloc((size_t)G4H_ * 4);
  float*    XgE     = (float*)alloc((size_t)BT_ * G4H_ * 4);
  float*    XgD     = (float*)alloc((size_t)BT_ * G4H_ * 4);
  ushort_t* hts_b   = (ushort_t*)alloc((size_t)BT_ * H_ * 2);
  ushort_t* proj_b  = (ushort_t*)alloc((size_t)BT_ * H_ * 2);
  ushort_t* hE0     = (ushort_t*)alloc((size_t)B_ * H_ * 2);
  ushort_t* hE1     = (ushort_t*)alloc((size_t)B_ * H_ * 2);
  float*    cE      = (float*)alloc((size_t)B_ * H_ * 4);
  ushort_t* hD0     = (ushort_t*)alloc((size_t)B_ * H_ * 2);
  ushort_t* hD1     = (ushort_t*)alloc((size_t)B_ * H_ * 2);
  float*    cD      = (float*)alloc((size_t)B_ * H_ * 4);
  float*    Gt      = (float*)alloc((size_t)B_ * G4H_ * 4);
  float*    logits  = (float*)alloc((size_t)B_ * T_ * 4);
  float*    aw      = (float*)alloc((size_t)B_ * T_ * 4);
  ushort_t* attn_b  = (ushort_t*)alloc((size_t)B_ * H_ * 2);
  ushort_t* attnp_b = (ushort_t*)alloc((size_t)B_ * H_ * 2);

  const dim3 blk256(256);
  const dim3 cvtGrid(2048);
  const float scale = 1.0f / 32.0f;   // 1/sqrt(H), H=1024

  // ---- precompute: bf16 conversions ----
  hipLaunchKernelGGL(cvt_f32_bf16, cvtGrid, blk256, 0, stream, x,       F_, xb,     F_, BT_,  F_);
  hipLaunchKernelGGL(cvt_f32_bf16, cvtGrid, blk256, 0, stream, enc_Wih, F_, Wih_b,  F_, G4H_, F_);
  hipLaunchKernelGGL(cvt_f32_bf16, cvtGrid, blk256, 0, stream, enc_Whh, H_, Whh_b,  H_, G4H_, H_);
  hipLaunchKernelGGL(cvt_f32_bf16, cvtGrid, blk256, 0, stream, We2d,    H_, We2d_b, H_, H_,   H_);
  hipLaunchKernelGGL(cvt_f32_bf16, cvtGrid, blk256, 0, stream, Wd2e,    H_, Wd2e_b, H_, H_,   H_);
  // dec_Wih is [4H, F+H]: split into x-part [4H,F] and attn-part [4H,H]
  hipLaunchKernelGGL(cvt_f32_bf16, cvtGrid, blk256, 0, stream, dec_Wih,      F_ + H_, dWx_b, F_, G4H_, F_);
  hipLaunchKernelGGL(cvt_f32_bf16, cvtGrid, blk256, 0, stream, dec_Wih + F_, F_ + H_, dWa_b, H_, G4H_, H_);
  hipLaunchKernelGGL(cvt_f32_bf16, cvtGrid, blk256, 0, stream, dec_Whh, H_, dWhh_b, H_, G4H_, H_);
  // fused biases
  hipLaunchKernelGGL(vec_add, dim3(G4H_ / 256), blk256, 0, stream, enc_bih, enc_bhh, encb, G4H_);
  hipLaunchKernelGGL(vec_add, dim3(G4H_ / 256), blk256, 0, stream, dec_bih, dec_bhh, decb, G4H_);
  // zero states
  hipLaunchKernelGGL(zero_u32, dim3(64), blk256, 0, stream, (unsigned int*)hE0, (long)B_ * H_ / 2);
  hipLaunchKernelGGL(zero_u32, dim3(64), blk256, 0, stream, (unsigned int*)hD0, (long)B_ * H_ / 2);
  hipLaunchKernelGGL(zero_u32, dim3(64), blk256, 0, stream, (unsigned int*)cE,  (long)B_ * H_);
  hipLaunchKernelGGL(zero_u32, dim3(64), blk256, 0, stream, (unsigned int*)cD,  (long)B_ * H_);

  // ---- big hoisted GEMMs: Xg = x @ Wih^T + b  (both enc and dec) ----
  hipLaunchKernelGGL(gemm_wmma_bf16, dim3(G4H_ / 128, BT_ / 64), blk256, 0, stream,
                     xb, F_, Wih_b, F_, (const float*)nullptr, 0, encb,
                     XgE, (ushort_t*)nullptr, G4H_, F_);
  hipLaunchKernelGGL(gemm_wmma_bf16, dim3(G4H_ / 128, BT_ / 64), blk256, 0, stream,
                     xb, F_, dWx_b, F_, (const float*)nullptr, 0, decb,
                     XgD, (ushort_t*)nullptr, G4H_, F_);

  // ---- encoder recurrence ----
  for (int t = 0; t < T_; ++t) {
    ushort_t* hc = (t & 1) ? hE1 : hE0;
    ushort_t* hn = (t & 1) ? hE0 : hE1;
    hipLaunchKernelGGL(gemm_wmma_bf16, dim3(G4H_ / 128, 1), blk256, 0, stream,
                       hc, H_, Whh_b, H_, XgE + (size_t)t * G4H_, T_ * G4H_,
                       (const float*)nullptr, Gt, (ushort_t*)nullptr, G4H_, H_);
    hipLaunchKernelGGL(lstm_cell_k, dim3(B_ * H_ / 256), blk256, 0, stream,
                       Gt, cE, hn, hts_b, t);
  }

  // ---- projection: proj = h_ts @ We2d^T + be2d  (bf16 out) ----
  hipLaunchKernelGGL(gemm_wmma_bf16, dim3(H_ / 128, BT_ / 64), blk256, 0, stream,
                     hts_b, H_, We2d_b, H_, (const float*)nullptr, 0, be2d,
                     (float*)nullptr, proj_b, H_, H_);

  // ---- decoder recurrence with attention ----
  for (int t = 0; t < T_; ++t) {
    ushort_t* hc = (t & 1) ? hD1 : hD0;
    ushort_t* hn = (t & 1) ? hD0 : hD1;
    hipLaunchKernelGGL(attn_logits_k, dim3(B_ * T_ * 32 / 256), blk256, 0, stream,
                       hc, proj_b, logits, scale);
    hipLaunchKernelGGL(softmax_rows_k, dim3(B_), dim3(T_), 0, stream,
                       logits, aw, (t == T_ - 1) ? out_aw : (float*)nullptr);
    hipLaunchKernelGGL(attn_sum_k, dim3(B_ * H_ / 256), blk256, 0, stream,
                       aw, proj_b, attn_b);
    // attnp = attn @ Wd2e^T + bd2e (bf16 out)
    hipLaunchKernelGGL(gemm_wmma_bf16, dim3(H_ / 128, 1), blk256, 0, stream,
                       attn_b, H_, Wd2e_b, H_, (const float*)nullptr, 0, bd2e,
                       (float*)nullptr, attnp_b, H_, H_);
    // G = XgD[t] + attnp @ dWa^T   (x-part + fused bias already in XgD)
    hipLaunchKernelGGL(gemm_wmma_bf16, dim3(G4H_ / 128, 1), blk256, 0, stream,
                       attnp_b, H_, dWa_b, H_, XgD + (size_t)t * G4H_, T_ * G4H_,
                       (const float*)nullptr, Gt, (ushort_t*)nullptr, G4H_, H_);
    // G += h @ dWhh^T
    hipLaunchKernelGGL(gemm_wmma_bf16, dim3(G4H_ / 128, 1), blk256, 0, stream,
                       hc, H_, dWhh_b, H_, Gt, G4H_,
                       (const float*)nullptr, Gt, (ushort_t*)nullptr, G4H_, H_);
    hipLaunchKernelGGL(lstm_cell_k, dim3(B_ * H_ / 256), blk256, 0, stream,
                       Gt, cD, hn, (ushort_t*)nullptr, t);
    hipLaunchKernelGGL(classifier_k, dim3((B_ * C_ + 255) / 256), blk256, 0, stream,
                       hn, Wc, bc, out_scores, t);
  }
}
